// VoxelResBackBone8xTS_81862076662466
// MI455X (gfx1250) — compile-verified, matrix operand order checked by reading, not run on hardware
//
#include <hip/hip_runtime.h>

// ---------------------------------------------------------------------------
// VoxelResBackBone8x on MI455X (gfx1250): implicit-GEMM 3D convs on WMMA.
// f16 storage / f16 WMMA / f32 accumulate. One wave32 = one 16x16 output tile.
// Workgroup = 8 waves sharing one output-channel tile; B fragments staged into
// LDS by the Tensor Data Mover (double-buffered), A gathered as b128 loads.
// ---------------------------------------------------------------------------

typedef __attribute__((ext_vector_type(16))) _Float16 v16h;
typedef __attribute__((ext_vector_type(8)))  _Float16 h8;
typedef __attribute__((ext_vector_type(8)))  float    v8f;
typedef unsigned int u32x4 __attribute__((ext_vector_type(4)));
typedef int          i32x8 __attribute__((ext_vector_type(8)));
typedef int          i32x4 __attribute__((ext_vector_type(4)));

#define EPS_BN 1e-3f

#if defined(__has_builtin)
#if __has_builtin(__builtin_amdgcn_tensor_load_to_lds) && \
    __has_builtin(__builtin_amdgcn_s_wait_tensorcnt)
#define HAVE_TDM 1
#endif
#endif
#ifndef HAVE_TDM
#define HAVE_TDM 0
#endif

#if HAVE_TDM
// ---------------------------------------------------------------------------
// TDM: DMA a 1KB (256 x 4B) block from global memory into LDS.
// D# group0: count=1, lds_addr, 57-bit global addr, type=2 ("image").
// D# group1: data_size=4B, tensor_dim0 = tile_dim0 = 256, stride 256; 1-D tile.
// ---------------------------------------------------------------------------
__device__ inline void tdm_load_1kb(unsigned ldsoff, unsigned long long gaddr)
{
    u32x4 g0 = {};
    g0[0] = 1u;                                     // count=1, user descriptor
    g0[1] = ldsoff;                                 // LDS byte address
    g0[2] = (unsigned)gaddr;                        // global_addr[31:0]
    g0[3] = (unsigned)((gaddr >> 32) & 0x01FFFFFFull) | (2u << 30); // [56:32], type=2
    i32x8 g1 = {};
    g1[0] = (int)(2u << 16);                        // data_size = 2 -> 4 bytes
    g1[1] = (int)(256u << 16);                      // tensor_dim0[15:0] = 256
    g1[2] = 0;                                      // tensor_dim0 hi / tensor_dim1 lo
    g1[3] = (int)(256u << 16);                      // tile_dim0 = 256
    g1[4] = 0;                                      // tile_dim1/2 unused (1-D)
    g1[5] = 256;                                    // tensor_dim0_stride = 256
    g1[6] = 0;
    g1[7] = 0;
    i32x4 g2 = {};
    i32x4 g3 = {};
#if __clang_major__ >= 23
    i32x8 g4 = {};
    __builtin_amdgcn_tensor_load_to_lds(g0, g1, g2, g3, g4, 0);
#else
    __builtin_amdgcn_tensor_load_to_lds(g0, g1, g2, g3, 0);
#endif
}
#endif

// ---------------------------------------------------------------------------
// Scatter sparse voxels into dense f16 NDHWC grid (Cin padded 5 -> 8) + mask.
// ---------------------------------------------------------------------------
__global__ __launch_bounds__(256) void scatter_voxels(
    const float* __restrict__ feats, const int* __restrict__ coords,
    _Float16* __restrict__ dense, _Float16* __restrict__ mask, int n)
{
    int i = blockIdx.x * blockDim.x + threadIdx.x;
    if (i >= n) return;
    int z = coords[i * 4 + 1];
    int y = coords[i * 4 + 2];
    int x = coords[i * 4 + 3];
    long vox = ((long)z * 200 + y) * 200 + x;
#pragma unroll
    for (int c = 0; c < 5; ++c)
        dense[vox * 8 + c] = (_Float16)feats[i * 5 + c];
    mask[vox] = (_Float16)1.0f;
}

// ---------------------------------------------------------------------------
// Pack DHWIO f32 weights into the exact B-fragment layout:
//   pk[((nt*KC + kc)*32 + lane)*16 + e],  K(e) = kc*32 + (lane>>4)*8 + (e&7) + ((e>>3)<<4)
// ---------------------------------------------------------------------------
__global__ __launch_bounds__(256) void prep_weights(
    const float* __restrict__ w, _Float16* __restrict__ pk,
    int taps, int cinr, int cinp, int cout)
{
    const int KTOT = taps * cinp;
    const int KC   = (KTOT + 31) >> 5;
    const int NT   = cout >> 4;
    const int total = NT * KC * 512;
    int idx = blockIdx.x * blockDim.x + threadIdx.x;
    if (idx >= total) return;
    int e    = idx & 15;
    int lane = (idx >> 4) & 31;
    int rest = idx >> 9;
    int kc   = rest % KC;
    int nt   = rest / KC;
    int K = (kc << 5) + ((lane >> 4) << 3) + (e & 7) + ((e >> 3) << 4);
    int n = (nt << 4) + (lane & 15);
    float val = 0.f;
    if (K < KTOT) {
        int tap = K / cinp;
        int ci  = K % cinp;
        if (ci < cinr)
            val = w[((long)tap * cinr + ci) * cout + n];
    }
    pk[idx] = (_Float16)val;
}

// ---------------------------------------------------------------------------
// Fold BN (+ optional conv bias) into per-channel scale/shift.
// ---------------------------------------------------------------------------
__global__ __launch_bounds__(64) void prep_bn(
    const float* __restrict__ g, const float* __restrict__ b,
    const float* __restrict__ m, const float* __restrict__ v,
    const float* __restrict__ bias,
    float* __restrict__ scale, float* __restrict__ shift, int C)
{
    int c = blockIdx.x * blockDim.x + threadIdx.x;
    if (c >= C) return;
    float s  = g[c] * rsqrtf(v[c] + EPS_BN);
    float sh = b[c] - m[c] * s;
    if (bias) sh += bias[c] * s;
    scale[c] = s;
    shift[c] = sh;
}

// ---------------------------------------------------------------------------
// Mask downsample: max-pool window > 0 ? 1 : 0
// ---------------------------------------------------------------------------
__global__ __launch_bounds__(256) void mask_down(
    const _Float16* __restrict__ mi, _Float16* __restrict__ mo,
    int Din, int Hin, int Win, int Dout, int Hout, int Wout,
    int kd, int kh, int kw, int sd, int sh, int sw, int pd, int ph, int pw)
{
    int v = blockIdx.x * blockDim.x + threadIdx.x;
    int nv = Dout * Hout * Wout;
    if (v >= nv) return;
    int oz = v / (Hout * Wout);
    int r  = v - oz * (Hout * Wout);
    int oy = r / Wout;
    int ox = r - oy * Wout;
    float mx = 0.f;
    for (int dz = 0; dz < kd; ++dz)
        for (int dy = 0; dy < kh; ++dy)
            for (int dx = 0; dx < kw; ++dx) {
                int iz = oz * sd + dz - pd;
                int iy = oy * sh + dy - ph;
                int ix = ox * sw + dx - pw;
                if ((unsigned)iz < (unsigned)Din && (unsigned)iy < (unsigned)Hin &&
                    (unsigned)ix < (unsigned)Win) {
                    float t = (float)mi[((long)iz * Hin + iy) * Win + ix];
                    mx = mx > t ? mx : t;
                }
            }
    mo[v] = (_Float16)(mx > 0.f ? 1.f : 0.f);
}

// ---------------------------------------------------------------------------
// Implicit-GEMM conv. Block = 8 waves sharing one nt; wave w owns m-tile
// (blockIdx/NT)*8+w. Per K-chunk: TDM stages 1KB of B into LDS (double
// buffered), each wave does 2x b128 A-gather + 2x ds b128 B-read + 1 WMMA.
// KD/KH/KW are compile-time so tap->(dz,dy,dx) uses constant divisions.
// ---------------------------------------------------------------------------
template <int KD, int KH, int KW, int CIN, int COUT, bool OUT_F32>
__global__ __launch_bounds__(256) void conv3d_wmma(
    const _Float16* __restrict__ in, const _Float16* __restrict__ mout,
    const _Float16* __restrict__ res, const _Float16* __restrict__ wpk,
    const float* __restrict__ bnscale, const float* __restrict__ bnshift,
    void* __restrict__ outp,
    int Din, int Hin, int Win, int Dout, int Hout, int Wout,
    int sd, int sh, int sw, int pd, int ph, int pw)
{
    constexpr int NT   = COUT / 16;
    constexpr int TAPS = KD * KH * KW;
    constexpr int KTOT = TAPS * CIN;
    constexpr int KC   = (KTOT + 31) / 32;
    constexpr int LOG2CIN = (CIN == 8) ? 3 : (CIN == 16) ? 4 : (CIN == 32) ? 5
                          : (CIN == 64) ? 6 : 7;
    const int nvox = Dout * Hout * Wout;

    const int wv   = threadIdx.x >> 5;
    const int lane = threadIdx.x & 31;
    const int nt   = (int)(blockIdx.x % NT);
    const int mt   = (int)(blockIdx.x / NT) * 8 + wv;

    // A-fragment row for this lane (ISA 16-bit A 16x32 layout)
    const int rowm = lane & 15;
    const int hi   = lane >> 4;
    const int vox  = mt * 16 + rowm;
    const bool rowok = vox < nvox;
    int oz = 0, oy = 0, ox = 0;
    if (rowok) {
        unsigned t = (unsigned)vox;
        oz = t / (unsigned)(Hout * Wout); t -= (unsigned)oz * (Hout * Wout);
        oy = t / (unsigned)Wout;          ox = t - (unsigned)oy * Wout;
    }
    const int izb = oz * sd - pd, iyb = oy * sh - ph, ixb = ox * sw - pw;

#if HAVE_TDM
    __shared__ _Float16 Bs[2][512];
    const unsigned long long wbase =
        (unsigned long long)(const void*)(wpk + (size_t)(nt * KC) * 512);
    if (wv == 0) {
        tdm_load_1kb((unsigned)(size_t)(void*)&Bs[0][0], wbase);
        __builtin_amdgcn_s_wait_tensorcnt(0);
    }
    __syncthreads();
#else
    const _Float16* wq = wpk + (size_t)(nt * KC) * 512 + lane * 16;
#endif

    v8f c = {};
    for (int kc = 0; kc < KC; ++kc) {
#if HAVE_TDM
        const int buf = kc & 1;
        if (wv == 0 && kc + 1 < KC)
            tdm_load_1kb((unsigned)(size_t)(void*)&Bs[buf ^ 1][0],
                         wbase + (unsigned long long)(kc + 1) * 1024ull);
#endif
        // ---- A gather: two 8-contiguous-K groups -> two b128 loads --------
        const int kbase = (kc << 5) + (hi << 3);
        h8 a0 = {}, a1 = {};
#pragma unroll
        for (int g = 0; g < 2; ++g) {
            const int K0 = kbase + (g << 4);
            h8 av = {};
            if (rowok && K0 < KTOT) {
                const unsigned tap = (unsigned)K0 >> LOG2CIN;   // const-shift
                const int ci  = K0 & (CIN - 1);
                const int dz  = (int)(tap / (unsigned)(KH * KW)); // const divs
                const int rr  = (int)(tap % (unsigned)(KH * KW));
                const int dy  = rr / KW;
                const int dx  = rr % KW;
                const int iz = izb + dz, iy = iyb + dy, ix = ixb + dx;
                if ((unsigned)iz < (unsigned)Din && (unsigned)iy < (unsigned)Hin &&
                    (unsigned)ix < (unsigned)Win) {
                    const size_t addr = (size_t)((iz * Hin + iy) * Win + ix) * CIN + ci;
                    av = *reinterpret_cast<const h8*>(in + addr);
                }
            }
            if (g == 0) a0 = av; else a1 = av;
        }
        // ---- B fragment -----------------------------------------------------
#if HAVE_TDM
        const _Float16* bq = &Bs[buf][lane * 16];       // ds_load_b128 x2
#else
        const _Float16* bq = wq;
        wq += 512;
        __builtin_prefetch(wq, 0, 3);
#endif
        h8 b0 = *reinterpret_cast<const h8*>(bq);
        h8 b1 = *reinterpret_cast<const h8*>(bq + 8);

        v16h A = __builtin_shufflevector(a0, a1, 0, 1, 2, 3, 4, 5, 6, 7,
                                         8, 9, 10, 11, 12, 13, 14, 15);
        v16h B = __builtin_shufflevector(b0, b1, 0, 1, 2, 3, 4, 5, 6, 7,
                                         8, 9, 10, 11, 12, 13, 14, 15);
        c = __builtin_amdgcn_wmma_f32_16x16x32_f16(false, A, false, B,
                                                   (short)0, c, false, false);
#if HAVE_TDM
        if (wv == 0 && kc + 1 < KC) __builtin_amdgcn_s_wait_tensorcnt(0);
        __syncthreads();
#endif
    }

    // ---- epilogue (C layout: lane<16 -> N=lane,M=r ; lane>=16 -> M=r+8) ----
    const int ch = nt * 16 + (lane & 15);
    const float sc  = bnscale[ch];
    const float shv = bnshift[ch];
#pragma unroll
    for (int r = 0; r < 8; ++r) {
        const int m  = r + (hi << 3);
        const int v2 = mt * 16 + m;
        if (v2 < nvox) {
            float t = c[r] * sc + shv;
            t *= (float)mout[v2];
            if (res) t += (float)res[(size_t)v2 * COUT + ch];
            t = t > 0.f ? t : 0.f;
            const size_t oidx = (size_t)v2 * COUT + ch;
            if (OUT_F32) ((float*)outp)[oidx] = t;
            else         ((_Float16*)outp)[oidx] = (_Float16)t;
        }
    }
}

// ---------------------------------------------------------------------------
// Host-side helpers
// ---------------------------------------------------------------------------
template <int KD, int KH, int KW, int CIN, int COUT, bool OF32>
static void launch_conv(hipStream_t s, const _Float16* in, const _Float16* mo,
                        const _Float16* res, const _Float16* wpk,
                        const float* sc, const float* sh, void* out,
                        int Di, int Hi, int Wi, int Do, int Ho, int Wo,
                        int sd, int sth, int sw, int pd, int ph, int pw)
{
    int nvox   = Do * Ho * Wo;
    int mtiles = (nvox + 15) / 16;
    int blocks = ((mtiles + 7) / 8) * (COUT / 16);
    conv3d_wmma<KD, KH, KW, CIN, COUT, OF32><<<blocks, 256, 0, s>>>(
        in, mo, res, wpk, sc, sh, out, Di, Hi, Wi, Do, Ho, Wo,
        sd, sth, sw, pd, ph, pw);
}

struct LayerMeta { int wi, bi, gi, bbi, mi, vi, kd, kh, kw, cinr, cinp, cout; };

extern "C" void kernel_launch(void* const* d_in, const int* in_sizes, int n_in,
                              void* d_out, int out_size, void* d_ws, size_t ws_size,
                              hipStream_t stream)
{
    (void)in_sizes; (void)n_in; (void)out_size; (void)ws_size;
    const float* feats = (const float*)d_in[0];
    const int*  coords = (const int*)d_in[1];

    // ---- bump allocator over workspace ------------------------------------
    char* base = (char*)d_ws;
    size_t off = 0;
    auto alloc = [&](size_t bytes) -> char* {
        char* p = base + off;
        off = (off + bytes + 255) & ~size_t(255);
        return p;
    };

    const int n0 = 32 * 200 * 200;
    const int n2 = 16 * 100 * 100;
    const int n3 = 8 * 50 * 50;
    const int n4 = 3 * 25 * 25;
    const int n5 = 1 * 25 * 25;

    _Float16* dense = (_Float16*)alloc((size_t)n0 * 8 * 2);
    _Float16* m0 = (_Float16*)alloc((size_t)n0 * 2);
    _Float16* m2 = (_Float16*)alloc((size_t)n2 * 2);
    _Float16* m3 = (_Float16*)alloc((size_t)n3 * 2);
    _Float16* m4 = (_Float16*)alloc((size_t)n4 * 2);
    _Float16* m5 = (_Float16*)alloc((size_t)n5 * 2);
    _Float16* x0a = (_Float16*)alloc((size_t)n0 * 16 * 2);
    _Float16* x0b = (_Float16*)alloc((size_t)n0 * 16 * 2);
    _Float16* x2a = (_Float16*)alloc((size_t)n2 * 32 * 2);
    _Float16* x2b = (_Float16*)alloc((size_t)n2 * 32 * 2);
    _Float16* x3a = (_Float16*)alloc((size_t)n3 * 64 * 2);
    _Float16* x3b = (_Float16*)alloc((size_t)n3 * 64 * 2);
    _Float16* x4a = (_Float16*)alloc((size_t)n4 * 128 * 2);
    _Float16* x4b = (_Float16*)alloc((size_t)n4 * 128 * 2);

    // ---- per-layer metadata (d_in indices follow setup_inputs dict order) --
    static const LayerMeta L[21] = {
        {  2, -1,   3,   4,   5,   6, 3, 3, 3,   5,   8,  16}, // conv_input
        {  7,  8,   9,  10,  11,  12, 3, 3, 3,  16,  16,  16}, // conv1 b1 c1
        { 13, 14,  15,  16,  17,  18, 3, 3, 3,  16,  16,  16}, // conv1 b1 c2
        { 19, 20,  21,  22,  23,  24, 3, 3, 3,  16,  16,  16}, // conv1 b2 c1
        { 25, 26,  27,  28,  29,  30, 3, 3, 3,  16,  16,  16}, // conv1 b2 c2
        { 31, -1,  32,  33,  34,  35, 3, 3, 3,  16,  16,  32}, // conv2 down
        { 36, 37,  38,  39,  40,  41, 3, 3, 3,  32,  32,  32}, // conv2 b1 c1
        { 42, 43,  44,  45,  46,  47, 3, 3, 3,  32,  32,  32}, // conv2 b1 c2
        { 48, 49,  50,  51,  52,  53, 3, 3, 3,  32,  32,  32}, // conv2 b2 c1
        { 54, 55,  56,  57,  58,  59, 3, 3, 3,  32,  32,  32}, // conv2 b2 c2
        { 60, -1,  61,  62,  63,  64, 3, 3, 3,  32,  32,  64}, // conv3 down
        { 65, 66,  67,  68,  69,  70, 3, 3, 3,  64,  64,  64}, // conv3 b1 c1
        { 71, 72,  73,  74,  75,  76, 3, 3, 3,  64,  64,  64}, // conv3 b1 c2
        { 77, 78,  79,  80,  81,  82, 3, 3, 3,  64,  64,  64}, // conv3 b2 c1
        { 83, 84,  85,  86,  87,  88, 3, 3, 3,  64,  64,  64}, // conv3 b2 c2
        { 89, -1,  90,  91,  92,  93, 3, 3, 3,  64,  64, 128}, // conv4 down
        { 94, 95,  96,  97,  98,  99, 3, 3, 3, 128, 128, 128}, // conv4 b1 c1
        {100,101, 102, 103, 104, 105, 3, 3, 3, 128, 128, 128}, // conv4 b1 c2
        {106,107, 108, 109, 110, 111, 3, 3, 3, 128, 128, 128}, // conv4 b2 c1
        {112,113, 114, 115, 116, 117, 3, 3, 3, 128, 128, 128}, // conv4 b2 c2
        {118, -1, 119, 120, 121, 122, 3, 1, 1, 128, 128, 128}, // conv_out
    };

    _Float16* wpk[21];
    float* bnsc[21];
    float* bnsh[21];
    for (int i = 0; i < 21; ++i) {
        const LayerMeta& M = L[i];
        int taps = M.kd * M.kh * M.kw;
        int KC   = (taps * M.cinp + 31) / 32;
        int pel  = (M.cout / 16) * KC * 512;
        wpk[i] = (_Float16*)alloc((size_t)pel * 2);
        float* bn = (float*)alloc((size_t)2 * M.cout * 4);
        bnsc[i] = bn;
        bnsh[i] = bn + M.cout;
        prep_weights<<<(pel + 255) / 256, 256, 0, stream>>>(
            (const float*)d_in[M.wi], wpk[i], taps, M.cinr, M.cinp, M.cout);
        prep_bn<<<(M.cout + 63) / 64, 64, 0, stream>>>(
            (const float*)d_in[M.gi], (const float*)d_in[M.bbi],
            (const float*)d_in[M.mi], (const float*)d_in[M.vi],
            M.bi >= 0 ? (const float*)d_in[M.bi] : nullptr,
            bnsc[i], bnsh[i], M.cout);
    }

    // ---- build dense input + mask -----------------------------------------
    hipMemsetAsync(dense, 0, (size_t)n0 * 8 * 2, stream);
    hipMemsetAsync(m0, 0, (size_t)n0 * 2, stream);
    scatter_voxels<<<(100000 + 255) / 256, 256, 0, stream>>>(feats, coords, dense, m0, 100000);

    // ---- conv_input + conv1 (2 residual blocks) @ (32,200,200) x16 --------
    launch_conv<3, 3, 3, 8, 16, false>(stream, dense, m0, nullptr, wpk[0], bnsc[0], bnsh[0], x0a,
                                       32, 200, 200, 32, 200, 200, 1, 1, 1, 1, 1, 1);
    launch_conv<3, 3, 3, 16, 16, false>(stream, x0a, m0, nullptr, wpk[1], bnsc[1], bnsh[1], x0b,
                                        32, 200, 200, 32, 200, 200, 1, 1, 1, 1, 1, 1);
    launch_conv<3, 3, 3, 16, 16, false>(stream, x0b, m0, x0a, wpk[2], bnsc[2], bnsh[2], x0a,
                                        32, 200, 200, 32, 200, 200, 1, 1, 1, 1, 1, 1);
    launch_conv<3, 3, 3, 16, 16, false>(stream, x0a, m0, nullptr, wpk[3], bnsc[3], bnsh[3], x0b,
                                        32, 200, 200, 32, 200, 200, 1, 1, 1, 1, 1, 1);
    launch_conv<3, 3, 3, 16, 16, false>(stream, x0b, m0, x0a, wpk[4], bnsc[4], bnsh[4], x0a,
                                        32, 200, 200, 32, 200, 200, 1, 1, 1, 1, 1, 1);

    // ---- conv2: down to (16,100,100) x32 + 2 blocks -----------------------
    mask_down<<<(n2 + 255) / 256, 256, 0, stream>>>(m0, m2, 32, 200, 200, 16, 100, 100,
                                                    3, 3, 3, 2, 2, 2, 1, 1, 1);
    launch_conv<3, 3, 3, 16, 32, false>(stream, x0a, m2, nullptr, wpk[5], bnsc[5], bnsh[5], x2a,
                                        32, 200, 200, 16, 100, 100, 2, 2, 2, 1, 1, 1);
    launch_conv<3, 3, 3, 32, 32, false>(stream, x2a, m2, nullptr, wpk[6], bnsc[6], bnsh[6], x2b,
                                        16, 100, 100, 16, 100, 100, 1, 1, 1, 1, 1, 1);
    launch_conv<3, 3, 3, 32, 32, false>(stream, x2b, m2, x2a, wpk[7], bnsc[7], bnsh[7], x2a,
                                        16, 100, 100, 16, 100, 100, 1, 1, 1, 1, 1, 1);
    launch_conv<3, 3, 3, 32, 32, false>(stream, x2a, m2, nullptr, wpk[8], bnsc[8], bnsh[8], x2b,
                                        16, 100, 100, 16, 100, 100, 1, 1, 1, 1, 1, 1);
    launch_conv<3, 3, 3, 32, 32, false>(stream, x2b, m2, x2a, wpk[9], bnsc[9], bnsh[9], x2a,
                                        16, 100, 100, 16, 100, 100, 1, 1, 1, 1, 1, 1);

    // ---- conv3: down to (8,50,50) x64 + 2 blocks --------------------------
    mask_down<<<(n3 + 255) / 256, 256, 0, stream>>>(m2, m3, 16, 100, 100, 8, 50, 50,
                                                    3, 3, 3, 2, 2, 2, 1, 1, 1);
    launch_conv<3, 3, 3, 32, 64, false>(stream, x2a, m3, nullptr, wpk[10], bnsc[10], bnsh[10], x3a,
                                        16, 100, 100, 8, 50, 50, 2, 2, 2, 1, 1, 1);
    launch_conv<3, 3, 3, 64, 64, false>(stream, x3a, m3, nullptr, wpk[11], bnsc[11], bnsh[11], x3b,
                                        8, 50, 50, 8, 50, 50, 1, 1, 1, 1, 1, 1);
    launch_conv<3, 3, 3, 64, 64, false>(stream, x3b, m3, x3a, wpk[12], bnsc[12], bnsh[12], x3a,
                                        8, 50, 50, 8, 50, 50, 1, 1, 1, 1, 1, 1);
    launch_conv<3, 3, 3, 64, 64, false>(stream, x3a, m3, nullptr, wpk[13], bnsc[13], bnsh[13], x3b,
                                        8, 50, 50, 8, 50, 50, 1, 1, 1, 1, 1, 1);
    launch_conv<3, 3, 3, 64, 64, false>(stream, x3b, m3, x3a, wpk[14], bnsc[14], bnsh[14], x3a,
                                        8, 50, 50, 8, 50, 50, 1, 1, 1, 1, 1, 1);

    // ---- conv4: down to (3,25,25) x128 (pad D=0) + 2 blocks ---------------
    mask_down<<<(n4 + 255) / 256, 256, 0, stream>>>(m3, m4, 8, 50, 50, 3, 25, 25,
                                                    3, 3, 3, 2, 2, 2, 0, 1, 1);
    launch_conv<3, 3, 3, 64, 128, false>(stream, x3a, m4, nullptr, wpk[15], bnsc[15], bnsh[15], x4a,
                                         8, 50, 50, 3, 25, 25, 2, 2, 2, 0, 1, 1);
    launch_conv<3, 3, 3, 128, 128, false>(stream, x4a, m4, nullptr, wpk[16], bnsc[16], bnsh[16], x4b,
                                          3, 25, 25, 3, 25, 25, 1, 1, 1, 1, 1, 1);
    launch_conv<3, 3, 3, 128, 128, false>(stream, x4b, m4, x4a, wpk[17], bnsc[17], bnsh[17], x4a,
                                          3, 25, 25, 3, 25, 25, 1, 1, 1, 1, 1, 1);
    launch_conv<3, 3, 3, 128, 128, false>(stream, x4a, m4, nullptr, wpk[18], bnsc[18], bnsh[18], x4b,
                                          3, 25, 25, 3, 25, 25, 1, 1, 1, 1, 1, 1);
    launch_conv<3, 3, 3, 128, 128, false>(stream, x4b, m4, x4a, wpk[19], bnsc[19], bnsh[19], x4a,
                                          3, 25, 25, 3, 25, 25, 1, 1, 1, 1, 1, 1);

    // ---- conv_out: (3,1,1) stride (2,1,1), no pad -> (1,25,25) x128 f32 ---
    mask_down<<<(n5 + 255) / 256, 256, 0, stream>>>(m4, m5, 3, 25, 25, 1, 25, 25,
                                                    3, 1, 1, 2, 1, 1, 0, 0, 0);
    launch_conv<3, 1, 1, 128, 128, true>(stream, x4a, m5, nullptr, wpk[20], bnsc[20], bnsh[20], d_out,
                                         3, 25, 25, 1, 25, 25, 2, 1, 1, 0, 0, 0);
}
